// DecoderLanguageModel_3058016715319
// MI455X (gfx1250) — matrix operand order repaired and checked by reading, not compiled, not run on hardware
//
#include <hip/hip_runtime.h>

typedef __attribute__((ext_vector_type(16))) __bf16 v16bf;
typedef __attribute__((ext_vector_type(8)))  float  v8f;
typedef int v4i __attribute__((vector_size(16)));

#if defined(__has_builtin)
#if __has_builtin(__builtin_amdgcn_global_load_async_to_lds_b128)
#define HAVE_ASYNC_B128 1
#endif
#if __has_builtin(__builtin_amdgcn_s_wait_asynccnt)
#define HAVE_WAIT_ASYNC 1
#endif
#endif

__device__ __forceinline__ void wait_async_zero() {
#if defined(HAVE_ASYNC_B128)
#if defined(HAVE_WAIT_ASYNC)
  __builtin_amdgcn_s_wait_asynccnt(0);
#else
  asm volatile("s_wait_asynccnt 0x0" ::: "memory");
#endif
#endif
}

// async 16-byte global -> LDS copy (ASYNCcnt-tracked); signature per hipcc:
// param1 = int4-vector pointer in AS(1), non-const.
__device__ __forceinline__ void async_copy16(const unsigned short* gsrc, unsigned short* ldst) {
#if defined(HAVE_ASYNC_B128)
  __builtin_amdgcn_global_load_async_to_lds_b128(
      (__attribute__((address_space(1))) v4i*)(unsigned short*)gsrc,
      (__attribute__((address_space(3))) v4i*)ldst, 0, 0);
#endif
}

union BF16x16 {
  v16bf v;
  unsigned int   u32[8];
  unsigned short u16[16];
};

__device__ __forceinline__ unsigned short f32_to_bf16(float f) {
  unsigned int u = __float_as_uint(f);
  unsigned int r = u + 0x7FFFu + ((u >> 16) & 1u);   // round-to-nearest-even
  return (unsigned short)(r >> 16);
}

// A-matrix (16x32 bf16) per-lane K offsets: K = 16*half + 2v (2 consecutive K per VGPR)
__device__ __forceinline__ int kofsA(int v, int kh) {
  return (kh << 4) + (v << 1);
}
// B-matrix (32x16 bf16): lanes 0-15 hold K=0..15, lanes 16-31 hold K=16..31, 2 consecutive K per VGPR
__device__ __forceinline__ int kofsB(int v, int kh) {
  return (kh << 4) + (v << 1);
}

// swizzled global index for a B-matrix element (k,n) of a [K,N] weight:
// fragments of 32 K x 16 N, 512 elems each, lane-major so a wave's fragment is
// 32 lanes x 16 contiguous ushorts (two coalesced b128 loads per lane).
__device__ __forceinline__ size_t bswz(int k, int n, int N) {
  const int kt = k >> 5, kk = k & 31;
  const int nt = n >> 4, nn = n & 15;
  const int lane = nn + ((kk >> 4) << 4);
  const int idx  = kk & 15;                 // 2*v + low-bit
  return ((size_t)kt * (size_t)(N >> 4) + nt) * 512 + (size_t)lane * 16 + idx;
}

__device__ __forceinline__ v8f zero8() {
  v8f z;
  #pragma unroll
  for (int i = 0; i < 8; ++i) z[i] = 0.0f;
  return z;
}

__device__ __forceinline__ v8f wmma_bf16(v16bf a, v16bf b, v8f c) {
  // (neg_a, A, neg_b, B, c_mod, C, reuse_a, reuse_b)
  return __builtin_amdgcn_wmma_f32_16x16x32_bf16(false, a, false, b, (short)0, c, false, false);
}

// ---------------------------------------------------------------------------
// f32 [L,K,N] weights -> bf16 in WMMA B-fragment swizzled layout (per slice)
// ---------------------------------------------------------------------------
__global__ __launch_bounds__(256) void cvt_swz_kernel(const float* __restrict__ in,
                                                      unsigned short* __restrict__ out,
                                                      int K, int N) {
  const size_t slice = (size_t)K * N;
  const float* src = in + slice * blockIdx.y;
  unsigned short* dst = out + slice * blockIdx.y;
  long long i = (long long)blockIdx.x * 256 + threadIdx.x;
  if (i < (long long)slice) {
    int k = (int)(i / N);
    int n = (int)(i - (long long)k * N);
    dst[bswz(k, n, N)] = f32_to_bf16(src[i]);
  }
}

// ---------------------------------------------------------------------------
// embedding * sqrt(D) + sinusoidal positional encoding  -> x (f32)
// ---------------------------------------------------------------------------
__global__ __launch_bounds__(256) void embed_kernel(const int* __restrict__ tokens,
                                                    const float* __restrict__ emb,
                                                    float* __restrict__ x,
                                                    int S, int Dd) {
  int idx = blockIdx.x * 256 + threadIdx.x;
  int n = idx / Dd;
  int d = idx - n * Dd;
  int s = n % S;
  int tok = tokens[n];
  float de  = (float)(d & ~1);
  float div = __expf(de * (-9.210340371976184f / (float)Dd));  // -ln(10000)/D
  float ang = (float)s * div;
  float pe  = (d & 1) ? __cosf(ang) : __sinf(ang);
  x[idx] = emb[(size_t)tok * Dd + d] * 27.712812921102035f + pe;  // sqrt(768)
}

// ---------------------------------------------------------------------------
// row LayerNorm: f32 in -> bf16 out
// ---------------------------------------------------------------------------
__global__ __launch_bounds__(256) void ln_kernel(const float* __restrict__ x,
                                                 const float* __restrict__ w,
                                                 const float* __restrict__ b,
                                                 unsigned short* __restrict__ out,
                                                 int Dd) {
  __shared__ float red[256];
  const int row = blockIdx.x;
  const int tid = threadIdx.x;
  const float* xr = x + (size_t)row * Dd;
  float s = 0.0f, s2 = 0.0f;
  for (int i = tid; i < Dd; i += 256) { float v = xr[i]; s += v; s2 += v * v; }
  red[tid] = s; __syncthreads();
  for (int o = 128; o > 0; o >>= 1) { if (tid < o) red[tid] += red[tid + o]; __syncthreads(); }
  float mean = red[0] / (float)Dd;
  __syncthreads();
  red[tid] = s2; __syncthreads();
  for (int o = 128; o > 0; o >>= 1) { if (tid < o) red[tid] += red[tid + o]; __syncthreads(); }
  float var = red[0] / (float)Dd - mean * mean;
  float inv = rsqrtf(var + 1e-5f);
  for (int i = tid; i < Dd; i += 256)
    out[(size_t)row * Dd + i] = f32_to_bf16((xr[i] - mean) * inv * w[i] + b[i]);
}

// ---------------------------------------------------------------------------
// bf16 WMMA GEMM:  C[M,N] = A[M,K] * Bswz[K,N] (+bias)(+resid)(relu)
// block = 256 threads (8 waves), tile 128x128, K-step 32; wave tile 32x64.
// A double-buffered in LDS (async global->LDS copies when available, register
// pipeline fallback); B fragments loaded coalesced from the pre-swizzled
// global layout; next-K B region prefetched (global_prefetch_b8).
// ---------------------------------------------------------------------------
template <bool HAS_BIAS, bool RELU, bool RESID, bool OUTBF, bool STREAM>
__global__ __launch_bounds__(256) void gemm_bf16_kernel(
    const unsigned short* __restrict__ A, const unsigned short* __restrict__ Bm,
    const float* __restrict__ bias, const float* __restrict__ resid,
    float* __restrict__ Cf, unsigned short* __restrict__ Cb,
    int M, int N, int Kd) {
  __shared__ unsigned short lA[2][128 * 40];   // 128 rows x 32 K (+8 pad), double buffered
  const int tid  = threadIdx.x;
  const int lane = tid & 31;
  const int wave = tid >> 5;
  const int wm = wave & 3;          // 4 waves along M
  const int wn = wave >> 2;         // 2 waves along N
  const int m0 = blockIdx.y * 128;
  const int n0 = blockIdx.x * 128;
  const int kh  = lane >> 4;
  const int l16 = lane & 15;
  const size_t nfr = (size_t)(N >> 4);   // fragment columns

  v8f acc[2][4];
  #pragma unroll
  for (int i = 0; i < 2; ++i)
    #pragma unroll
    for (int j = 0; j < 4; ++j) acc[i][j] = zero8();

  const int amr = tid >> 1;           // A tile row (0..127)
  const int ac0 = (tid & 1) << 4;     // A tile col (0 or 16)

  // prologue: stage first A tile
  {
    const unsigned short* asrc = A + (size_t)(m0 + amr) * Kd + ac0;
    unsigned short* adst = &lA[0][amr * 40 + ac0];
#if defined(HAVE_ASYNC_B128)
    async_copy16(asrc, adst);
    async_copy16(asrc + 8, adst + 8);
    wait_async_zero();
#else
    *(uint4*)adst       = ((const uint4*)asrc)[0];
    *(uint4*)(adst + 8) = ((const uint4*)asrc)[1];
#endif
  }
  __syncthreads();

  int cur = 0;
  for (int k0 = 0; k0 < Kd; k0 += 32) {
    const bool more = (k0 + 32) < Kd;

#if defined(HAVE_ASYNC_B128)
    // kick off next A tile as async global->LDS copies (no VGPR staging)
    if (more) {
      const unsigned short* asrc = A + (size_t)(m0 + amr) * Kd + (k0 + 32) + ac0;
      unsigned short* adst = &lA[cur ^ 1][amr * 40 + ac0];
      async_copy16(asrc, adst);
      async_copy16(asrc + 8, adst + 8);
      __builtin_prefetch(Bm + (((size_t)((k0 >> 5) + 1)) * nfr + (n0 >> 4)) * 512 + (size_t)tid * 16, 0, 1);
    }
#else
    // register-pipelined staging fallback
    uint4 na0 = {0, 0, 0, 0}, na1 = {0, 0, 0, 0};
    if (more) {
      const uint4* asrc = (const uint4*)(A + (size_t)(m0 + amr) * Kd + (k0 + 32) + ac0);
      na0 = asrc[0];
      na1 = asrc[1];
      __builtin_prefetch(Bm + (((size_t)((k0 >> 5) + 1)) * nfr + (n0 >> 4)) * 512 + (size_t)tid * 16, 0, 1);
    }
#endif

    // A fragments from LDS
    BF16x16 af[2];
    #pragma unroll
    for (int mt = 0; mt < 2; ++mt)
      #pragma unroll
      for (int vv = 0; vv < 8; ++vv)
        af[mt].u32[vv] = *(const unsigned int*)&lA[cur][(wm * 32 + mt * 16 + l16) * 40 + kofsA(vv, kh)];

    // B fragments: two coalesced b128 loads each, straight from global (L2-resident)
    const unsigned short* bbase =
        Bm + ((size_t)(k0 >> 5) * nfr + ((n0 + wn * 64) >> 4)) * 512 + (size_t)lane * 16;
    #pragma unroll
    for (int nt = 0; nt < 4; ++nt) {
      BF16x16 bfr;
      *(uint4*)&bfr.u32[0] = *(const uint4*)(bbase + nt * 512);
      *(uint4*)&bfr.u32[4] = *(const uint4*)(bbase + nt * 512 + 8);
      acc[0][nt] = wmma_bf16(af[0].v, bfr.v, acc[0][nt]);
      acc[1][nt] = wmma_bf16(af[1].v, bfr.v, acc[1][nt]);
    }

#if defined(HAVE_ASYNC_B128)
    if (more) wait_async_zero();   // own async copies landed; barrier covers the rest
#else
    if (more) {
      *(uint4*)&lA[cur ^ 1][amr * 40 + ac0]     = na0;
      *(uint4*)&lA[cur ^ 1][amr * 40 + ac0 + 8] = na1;
    }
#endif
    __syncthreads();
    cur ^= 1;
  }

  #pragma unroll
  for (int mt = 0; mt < 2; ++mt)
    #pragma unroll
    for (int nt = 0; nt < 4; ++nt)
      #pragma unroll
      for (int r = 0; r < 8; ++r) {
        const int m = m0 + wm * 32 + mt * 16 + r + kh * 8;
        const int n = n0 + wn * 64 + nt * 16 + l16;
        float val = acc[mt][nt][r];
        if (HAS_BIAS) val += bias[n];
        if (RESID)    val += resid[(size_t)m * N + n];
        if (RELU)     val  = fmaxf(val, 0.0f);
        if (OUTBF) {
          Cb[(size_t)m * N + n] = f32_to_bf16(val);
        } else if (STREAM) {
          __builtin_nontemporal_store(val, &Cf[(size_t)m * N + n]);
        } else {
          Cf[(size_t)m * N + n] = val;
        }
      }
}

// ---------------------------------------------------------------------------
// causal flash attention, bf16 Q/K/V [B*S, H*64] row-major, bf16 out.
// block = 128 threads (4 waves), 64 query rows/block (16 per wave), DH = 64.
// ---------------------------------------------------------------------------
#define ATT_PPITCH 34

__global__ __launch_bounds__(128) void attn_kernel(
    const unsigned short* __restrict__ Q, const unsigned short* __restrict__ K,
    const unsigned short* __restrict__ Vm, unsigned short* __restrict__ O,
    int S, int Dd) {
  const int b  = blockIdx.z;
  const int h  = blockIdx.y;
  const int q0 = blockIdx.x * 64;
  const int lane = threadIdx.x & 31;
  const int w    = threadIdx.x >> 5;
  const int kh   = lane >> 4;
  const int l16  = lane & 15;
  const int hc   = h * 64;
  const size_t rowbase = (size_t)b * S;

  __shared__ unsigned short lP[4][16 * ATT_PPITCH];

  // Q fragments: A-matrix 16 queries x 64 dh (two k-steps of 32), kept in regs
  BF16x16 aq[2];
  {
    const int qrow = q0 + w * 16 + l16;
    const unsigned short* qp = Q + (rowbase + qrow) * Dd + hc;
    #pragma unroll
    for (int ks = 0; ks < 2; ++ks)
      #pragma unroll
      for (int vv = 0; vv < 8; ++vv)
        aq[ks].u32[vv] = *(const unsigned int*)(qp + ks * 32 + kofsA(vv, kh));
  }

  v8f oacc[4];
  #pragma unroll
  for (int i = 0; i < 4; ++i) oacc[i] = zero8();
  float mrow[8], lrow[8];
  #pragma unroll
  for (int r = 0; r < 8; ++r) { mrow[r] = -3.0e38f; lrow[r] = 0.0f; }

  const int nkt = q0 / 32 + 2;   // causal: keys up to q0+63
  for (int kt = 0; kt < nkt; ++kt) {
    const int kb = kt * 32;

    // scores = Q * K^T : 16 x 32 per wave (two 16-key n-tiles)
    v8f sacc[2];
    #pragma unroll
    for (int nt = 0; nt < 2; ++nt) {
      sacc[nt] = zero8();
      #pragma unroll
      for (int ks = 0; ks < 2; ++ks) {
        BF16x16 bkf;
        const int key = kb + nt * 16 + l16;
        const unsigned short* kp = K + (rowbase + key) * Dd + hc + ks * 32;
        #pragma unroll
        for (int vv = 0; vv < 8; ++vv)
          bkf.u32[vv] = *(const unsigned int*)(kp + kofsB(vv, kh));
        sacc[nt] = wmma_bf16(aq[ks].v, bkf.v, sacc[nt]);
      }
    }

    // causal mask + online softmax (row stats via half-wave shuffles)
    float p0[8], p1[8], corr[8];
    #pragma unroll
    for (int r = 0; r < 8; ++r) {
      const int myrow = q0 + w * 16 + r + kh * 8;
      float s0 = sacc[0][r] * 0.125f;          // 1/sqrt(64)
      float s1 = sacc[1][r] * 0.125f;
      if (kb + l16 > myrow)      s0 = -3.0e38f;
      if (kb + 16 + l16 > myrow) s1 = -3.0e38f;
      float tmax = fmaxf(s0, s1);
      #pragma unroll
      for (int off = 8; off >= 1; off >>= 1)
        tmax = fmaxf(tmax, __shfl_xor(tmax, off, 32));
      const float mnew = fmaxf(mrow[r], tmax);
      const float c  = __expf(mrow[r] - mnew);
      const float e0 = __expf(s0 - mnew);
      const float e1 = __expf(s1 - mnew);
      float rs = e0 + e1;
      #pragma unroll
      for (int off = 8; off >= 1; off >>= 1)
        rs += __shfl_xor(rs, off, 32);
      lrow[r] = lrow[r] * c + rs;
      mrow[r] = mnew;
      corr[r] = c;
      p0[r] = e0; p1[r] = e1;
    }
    #pragma unroll
    for (int nt = 0; nt < 4; ++nt)
      #pragma unroll
      for (int r = 0; r < 8; ++r) oacc[nt][r] *= corr[r];

    // transpose P (C-layout) through LDS into an A-fragment
    __syncthreads();
    #pragma unroll
    for (int r = 0; r < 8; ++r) {
      const int prow = r + kh * 8;
      lP[w][prow * ATT_PPITCH + l16]      = f32_to_bf16(p0[r]);
      lP[w][prow * ATT_PPITCH + 16 + l16] = f32_to_bf16(p1[r]);
    }
    __syncthreads();
    BF16x16 ap;
    #pragma unroll
    for (int vv = 0; vv < 8; ++vv)
      ap.u32[vv] = *(const unsigned int*)&lP[w][l16 * ATT_PPITCH + kofsA(vv, kh)];

    // O += P * V : four 16-wide dh tiles
    #pragma unroll
    for (int nt = 0; nt < 4; ++nt) {
      BF16x16 bvf;
      #pragma unroll
      for (int vv = 0; vv < 8; ++vv) {
        const int kk = kofsB(vv, kh);
        bvf.u16[2 * vv]     = Vm[(rowbase + kb + kk) * Dd + hc + nt * 16 + l16];
        bvf.u16[2 * vv + 1] = Vm[(rowbase + kb + kk + 1) * Dd + hc + nt * 16 + l16];
      }
      oacc[nt] = wmma_bf16(ap.v, bvf.v, oacc[nt]);
    }
  }

  // normalize and store
  #pragma unroll
  for (int nt = 0; nt < 4; ++nt)
    #pragma unroll
    for (int r = 0; r < 8; ++r) {
      const int row = q0 + w * 16 + r + kh * 8;
      O[(rowbase + row) * Dd + hc + nt * 16 + l16] = f32_to_bf16(oacc[nt][r] / lrow[r]);
    }
}

// ---------------------------------------------------------------------------
// host orchestration
// ---------------------------------------------------------------------------
extern "C" void kernel_launch(void* const* d_in, const int* in_sizes, int n_in,
                              void* d_out, int out_size, void* d_ws, size_t ws_size,
                              hipStream_t stream) {
  (void)in_sizes; (void)n_in; (void)out_size; (void)ws_size;
  constexpr int Vv = 32000, Dd = 768, Hh = 12, Ll = 6, FFd = 3072, Bb = 2, Ss = 2048;
  constexpr int NT = Bb * Ss;   // 4096 token rows

  const int*   tokens = (const int*)  d_in[0];
  const float* emb    = (const float*)d_in[1];
  const float* Wq     = (const float*)d_in[2];
  const float* bq     = (const float*)d_in[3];
  const float* Wk     = (const float*)d_in[4];
  const float* bk_    = (const float*)d_in[5];
  const float* Wv     = (const float*)d_in[6];
  const float* bv_    = (const float*)d_in[7];
  const float* Wo     = (const float*)d_in[8];
  const float* ln1w   = (const float*)d_in[9];
  const float* ln1b   = (const float*)d_in[10];
  const float* ln2w   = (const float*)d_in[11];
  const float* ln2b   = (const float*)d_in[12];
  const float* W1     = (const float*)d_in[13];
  const float* b1     = (const float*)d_in[14];
  const float* W2     = (const float*)d_in[15];
  const float* b2     = (const float*)d_in[16];
  const float* lnfw   = (const float*)d_in[17];
  const float* lnfb   = (const float*)d_in[18];
  const float* Wout   = (const float*)d_in[19];
  const float* bout   = (const float*)d_in[20];

  char* base = (char*)d_ws;
  size_t off = 0;
  auto alloc = [&](size_t bytes) -> void* {
    void* p = base + off;
    off += (bytes + 255) & ~(size_t)255;
    return p;
  };

  const size_t DD = (size_t)Dd * Dd, DF = (size_t)Dd * FFd;
  unsigned short* WqB   = (unsigned short*)alloc((size_t)Ll * DD * 2);
  unsigned short* WkB   = (unsigned short*)alloc((size_t)Ll * DD * 2);
  unsigned short* WvB   = (unsigned short*)alloc((size_t)Ll * DD * 2);
  unsigned short* WoB   = (unsigned short*)alloc((size_t)Ll * DD * 2);
  unsigned short* W1B   = (unsigned short*)alloc((size_t)Ll * DF * 2);
  unsigned short* W2B   = (unsigned short*)alloc((size_t)Ll * DF * 2);
  unsigned short* WoutB = (unsigned short*)alloc((size_t)Dd * Vv * 2);
  float*          x     = (float*)alloc((size_t)NT * Dd * 4);
  unsigned short* hB    = (unsigned short*)alloc((size_t)NT * Dd * 2);
  unsigned short* qB    = (unsigned short*)alloc((size_t)NT * Dd * 2);
  unsigned short* kB    = (unsigned short*)alloc((size_t)NT * Dd * 2);
  unsigned short* vB    = (unsigned short*)alloc((size_t)NT * Dd * 2);
  unsigned short* aoB   = (unsigned short*)alloc((size_t)NT * Dd * 2);
  unsigned short* ffB   = (unsigned short*)alloc((size_t)NT * FFd * 2);

  // swizzled bf16 weight conversion: grid.y = number of layer slices
  auto cvtswz = [&](const float* src, unsigned short* dst, int K, int N, int slices) {
    const long long slice = (long long)K * N;
    dim3 g((unsigned)((slice + 255) / 256), (unsigned)slices);
    cvt_swz_kernel<<<g, 256, 0, stream>>>(src, dst, K, N);
  };
  cvtswz(Wq,   WqB,   Dd,  Dd, Ll);
  cvtswz(Wk,   WkB,   Dd,  Dd, Ll);
  cvtswz(Wv,   WvB,   Dd,  Dd, Ll);
  cvtswz(Wo,   WoB,   Dd,  Dd, Ll);
  cvtswz(W1,   W1B,   Dd,  FFd, Ll);
  cvtswz(W2,   W2B,   FFd, Dd, Ll);
  cvtswz(Wout, WoutB, Dd,  Vv, 1);

  embed_kernel<<<(NT * Dd) / 256, 256, 0, stream>>>(tokens, emb, x, Ss, Dd);

  const dim3 gD(Dd / 128, NT / 128);     // D-wide GEMMs
  const dim3 gF(FFd / 128, NT / 128);    // FF-wide GEMM
  for (int l = 0; l < Ll; ++l) {
    ln_kernel<<<NT, 256, 0, stream>>>(x, ln1w + (size_t)l * Dd, ln1b + (size_t)l * Dd, hB, Dd);
    gemm_bf16_kernel<true, false, false, true, false><<<gD, 256, 0, stream>>>(
        hB, WqB + (size_t)l * DD, bq + (size_t)l * Dd, nullptr, nullptr, qB, NT, Dd, Dd);
    gemm_bf16_kernel<true, false, false, true, false><<<gD, 256, 0, stream>>>(
        hB, WkB + (size_t)l * DD, bk_ + (size_t)l * Dd, nullptr, nullptr, kB, NT, Dd, Dd);
    gemm_bf16_kernel<true, false, false, true, false><<<gD, 256, 0, stream>>>(
        hB, WvB + (size_t)l * DD, bv_ + (size_t)l * Dd, nullptr, nullptr, vB, NT, Dd, Dd);
    attn_kernel<<<dim3(Ss / 64, Hh, Bb), 128, 0, stream>>>(qB, kB, vB, aoB, Ss, Dd);
    gemm_bf16_kernel<false, false, true, false, false><<<gD, 256, 0, stream>>>(
        aoB, WoB + (size_t)l * DD, nullptr, x, x, nullptr, NT, Dd, Dd);
    ln_kernel<<<NT, 256, 0, stream>>>(x, ln2w + (size_t)l * Dd, ln2b + (size_t)l * Dd, hB, Dd);
    gemm_bf16_kernel<true, true, false, true, false><<<gF, 256, 0, stream>>>(
        hB, W1B + (size_t)l * DF, b1 + (size_t)l * FFd, nullptr, nullptr, ffB, NT, FFd, Dd);
    gemm_bf16_kernel<true, false, true, false, false><<<gD, 256, 0, stream>>>(
        ffB, W2B + (size_t)l * DF, b2 + (size_t)l * Dd, x, x, nullptr, NT, Dd, FFd);
  }

  ln_kernel<<<NT, 256, 0, stream>>>(x, lnfw, lnfb, hB, Dd);
  gemm_bf16_kernel<true, false, false, false, true><<<dim3(Vv / 128, NT / 128), 256, 0, stream>>>(
      hB, WoutB, bout, nullptr, (float*)d_out, nullptr, NT, Vv, Dd);
}
